// Interaction_5506148073800
// MI455X (gfx1250) — matrix operand order repaired and checked by reading, not compiled, not run on hardware
//
#include <hip/hip_runtime.h>
#include <stdint.h>

typedef __attribute__((ext_vector_type(16))) _Float16 v16h;
typedef __attribute__((ext_vector_type(8)))  _Float16 v8h;
typedef __attribute__((ext_vector_type(4)))  _Float16 v4h;
typedef __attribute__((ext_vector_type(8)))  float    v8f;
typedef __attribute__((ext_vector_type(4)))  unsigned int v4u;
typedef __attribute__((ext_vector_type(8)))  int      v8i;
typedef __attribute__((ext_vector_type(4)))  int      v4i;

#define LOG2F 0.69314718055994530942f

__device__ __forceinline__ float sspf(float x) {
    // softplus(x) - log(2), numerically stable
    return fmaxf(x, 0.0f) + __logf(1.0f + __expf(-fabsf(x))) - LOG2F;
}

__device__ __forceinline__ v8f wmma16(v16h a, v16h b, v8f c) {
    return __builtin_amdgcn_wmma_f32_16x16x32_f16(
        /*neg_a=*/false, a, /*neg_b=*/false, b,
        /*c_mod=*/(short)0, c, /*reuse_a=*/false, /*reuse_b=*/false);
}

__device__ __forceinline__ v16h make_a(v8h lo, v8h hi) {
    v16h a;
    #pragma unroll
    for (int j = 0; j < 8; ++j) { a[j] = lo[j]; a[8 + j] = hi[j]; }
    return a;
}

// ---------------------------------------------------------------------------
// Pack a row-major [K x 64] f32 weight matrix into f16 B-operand fragments.
// Per ISA 7.12.2 (16-bit B 32x16): lane holds col = lane&15, K-halves
// j=0..15 are K = (lane<16 ? 0 : 8) + j + (j<8 ? 0 : 8).  dst[(f*32 + lane)*16 + j].
// ---------------------------------------------------------------------------
__global__ void pack_b(const float* __restrict__ W, _Float16* __restrict__ dst,
                       int Kreal, int nfrag) {
    int idx = blockIdx.x * blockDim.x + threadIdx.x;
    int total = nfrag * 512;
    if (idx >= total) return;
    int f    = idx >> 9;
    int w    = idx & 511;
    int lane = w >> 4;
    int j    = w & 15;
    int kstep = f >> 2, nt = f & 3;
    int col = nt * 16 + (lane & 15);
    int kb  = (lane < 16) ? 0 : 8;
    int k   = kstep * 32 + kb + j + ((j >= 8) ? 8 : 0);
    float v = (k < Kreal) ? W[k * 64 + col] : 0.0f;
    dst[idx] = (_Float16)v;
}

__global__ void zero_f32(float* __restrict__ p, int n4) {
    int i = blockIdx.x * blockDim.x + threadIdx.x;
    if (i < n4) {
        float4 z; z.x = 0.f; z.y = 0.f; z.z = 0.f; z.w = 0.f;
        ((float4*)p)[i] = z;
    }
}

// ---------------------------------------------------------------------------
// h = x @ w1 + b1.  One wave per 16-row tile; K=64 (2 ksteps), 4 N-tiles.
// ---------------------------------------------------------------------------
__global__ void __launch_bounds__(128)
node_linear(const float* __restrict__ x, const _Float16* __restrict__ w1p,
            const float* __restrict__ b1, float* __restrict__ h, int N) {
    int lane = threadIdx.x & 31, wv = threadIdx.x >> 5;
    int n_lo = lane & 15, hi = lane >> 4;
    int kbA = hi * 8, rb = hi * 8;
    int tiles = (N + 15) >> 4;
    for (int t = blockIdx.x * 4 + wv; t < tiles; t += gridDim.x * 4) {
        int r0 = t << 4;
        int rowc = min(r0 + n_lo, N - 1);
        const float* xr = x + (size_t)rowc * 64;
        v8f acc[4] = {};
        #pragma unroll
        for (int ks = 0; ks < 2; ++ks) {
            int kb = ks * 32 + kbA;
            float4 c0 = *(const float4*)(xr + kb);
            float4 c1 = *(const float4*)(xr + kb + 4);
            float4 c2 = *(const float4*)(xr + kb + 16);
            float4 c3 = *(const float4*)(xr + kb + 20);
            v16h a;
            a[0]=(_Float16)c0.x;  a[1]=(_Float16)c0.y;  a[2]=(_Float16)c0.z;  a[3]=(_Float16)c0.w;
            a[4]=(_Float16)c1.x;  a[5]=(_Float16)c1.y;  a[6]=(_Float16)c1.z;  a[7]=(_Float16)c1.w;
            a[8]=(_Float16)c2.x;  a[9]=(_Float16)c2.y;  a[10]=(_Float16)c2.z; a[11]=(_Float16)c2.w;
            a[12]=(_Float16)c3.x; a[13]=(_Float16)c3.y; a[14]=(_Float16)c3.z; a[15]=(_Float16)c3.w;
            #pragma unroll
            for (int nt = 0; nt < 4; ++nt) {
                v16h b = *(const v16h*)(w1p + (size_t)((ks * 4 + nt) * 32 + lane) * 16);
                acc[nt] = wmma16(a, b, acc[nt]);
            }
        }
        if (r0 + 16 <= N) {                 // full tile: branch-free stores
            #pragma unroll
            for (int nt = 0; nt < 4; ++nt) {
                int col = nt * 16 + n_lo;
                float bb = b1[col];
                #pragma unroll
                for (int v = 0; v < 8; ++v)
                    h[(size_t)(r0 + rb + v) * 64 + col] = acc[nt][v] + bb;
            }
        } else {
            #pragma unroll
            for (int nt = 0; nt < 4; ++nt) {
                int col = nt * 16 + n_lo;
                float bb = b1[col];
                #pragma unroll
                for (int v = 0; v < 8; ++v) {
                    int rr = r0 + rb + v;
                    if (rr < N) h[(size_t)rr * 64 + col] = acc[nt][v] + bb;
                }
            }
        }
    }
}

// ---------------------------------------------------------------------------
// CFConv edge kernel.  fw1 fragments (40 KB) are DMA'd into LDS once per
// block by the Tensor Data Mover (tensor_load_to_lds + s_wait_tensorcnt).
// ---------------------------------------------------------------------------
__global__ void __launch_bounds__(128)
edge_cfconv(const float* __restrict__ rbf, const int* __restrict__ eidx,
            const float* __restrict__ h,
            const _Float16* __restrict__ fw1p, const float* __restrict__ fb1,
            const _Float16* __restrict__ fw2p, const float* __restrict__ fb2,
            float* __restrict__ out, int E) {
    __shared__ _Float16 sFW1[20480];      // 40 frags * 512 halves = 40 KB
    __shared__ _Float16 sBuf[4][1152];    // per-wave staging / transpose

    // --- TDM: DMA fw1p (40960 B = 10240 dwords) into LDS, wave 0 only ---
    if ((threadIdx.x >> 5) == 0) {
        unsigned long long ga = (unsigned long long)(uintptr_t)fw1p;
        unsigned int lds = (unsigned int)(uintptr_t)&sFW1[0];
        v4u g0;
        g0.x = 1u;                                           // count=1, user D#
        g0.y = lds;                                          // lds_addr
        g0.z = (unsigned int)(ga & 0xffffffffu);             // global_addr lo
        g0.w = (unsigned int)((ga >> 32) & 0x01ffffffu) | (2u << 30);  // hi | type=2
        v8i g1;
        g1[0] = (2 << 16);                    // workgroup_mask=0, data_size=4B
        g1[1] = (int)(10240u << 16);          // tensor_dim0[15:0] -> bits 63:48
        g1[2] = (int)(1u << 16);              // tensor_dim0 hi=0, tensor_dim1=1
        g1[3] = (int)(10240u << 16);          // tile_dim0 = 10240 dwords
        g1[4] = 1;                            // tile_dim1 = 1, tile_dim2 = 0
        g1[5] = 10240;                        // tensor_dim0_stride (dwords)
        g1[6] = 0;
        g1[7] = 0;
        v4i g2 = {0, 0, 0, 0};
        v4i g3 = {0, 0, 0, 0};
        v8i g4 = {0, 0, 0, 0, 0, 0, 0, 0};    // 6-arg toolchain: extra group
        __builtin_amdgcn_tensor_load_to_lds(g0, g1, g2, g3, g4, 0);
        __builtin_amdgcn_s_wait_tensorcnt(0);
    }
    __syncthreads();

    int lane = threadIdx.x & 31, wv = threadIdx.x >> 5;
    int n_lo = lane & 15, hi = lane >> 4;
    int kbA = hi * 8, rb = hi * 8;
    _Float16* stg = sBuf[wv];             // [16][40] halves (80B rows, 16B aligned)
    _Float16* tb  = sBuf[wv];             // [16][72] halves (144B rows, 16B aligned)
    int sr = lane >> 1, sc = (lane & 1) * 16;

    int tiles = (E + 15) >> 4;
    for (int t = blockIdx.x * 4 + wv; t < tiles; t += gridDim.x * 4) {
        int e0 = t << 4;
        int rowc = min(e0 + sr, E - 1);
        const float* gpr = rbf + (size_t)rowc * 300 + sc;

        // ---- GEMM1: [16 x 320(pad)] x [320 x 64] ----
        v8f acc[4] = {};
        for (int ks = 0; ks < 9; ++ks) {          // full K-steps, branch-free
            #pragma unroll
            for (int j = 0; j < 16; j += 4) {
                float4 v = *(const float4*)(gpr + ks * 32 + j);
                v4h hh; hh[0]=(_Float16)v.x; hh[1]=(_Float16)v.y;
                        hh[2]=(_Float16)v.z; hh[3]=(_Float16)v.w;
                *(v4h*)(stg + sr * 40 + sc + j) = hh;
            }
            __builtin_prefetch(gpr + (ks + 2) * 32, 0, 1);   // global_prefetch_b8
            asm volatile("" ::: "memory");
            v8h lo = *(const v8h*)(stg + n_lo * 40 + kbA);
            v8h hx = *(const v8h*)(stg + n_lo * 40 + 16 + kbA);
            v16h a = make_a(lo, hx);
            #pragma unroll
            for (int nt = 0; nt < 4; ++nt) {
                v16h b = *(const v16h*)(sFW1 + (size_t)((ks * 4 + nt) * 32 + lane) * 16);
                acc[nt] = wmma16(a, b, acc[nt]);
            }
            asm volatile("" ::: "memory");
        }
        {   // K-tail: ks = 9 covers K=288..319, only K<300 is real
            #pragma unroll
            for (int j = 0; j < 16; j += 4) {
                int kk = 288 + sc + j;
                float4 v; v.x = 0.f; v.y = 0.f; v.z = 0.f; v.w = 0.f;
                if (kk < 300) v = *(const float4*)(gpr + 288 + j);
                v4h hh; hh[0]=(_Float16)v.x; hh[1]=(_Float16)v.y;
                        hh[2]=(_Float16)v.z; hh[3]=(_Float16)v.w;
                *(v4h*)(stg + sr * 40 + sc + j) = hh;
            }
            asm volatile("" ::: "memory");
            v8h lo = *(const v8h*)(stg + n_lo * 40 + kbA);
            v8h hx = *(const v8h*)(stg + n_lo * 40 + 16 + kbA);
            v16h a = make_a(lo, hx);
            #pragma unroll
            for (int nt = 0; nt < 4; ++nt) {
                v16h b = *(const v16h*)(sFW1 + (size_t)((9 * 4 + nt) * 32 + lane) * 16);
                acc[nt] = wmma16(a, b, acc[nt]);
            }
            asm volatile("" ::: "memory");
        }

        // bias + ssp, write C-layout -> A-layout transpose buffer (f16)
        #pragma unroll
        for (int nt = 0; nt < 4; ++nt) {
            int col = nt * 16 + n_lo;
            float bb = fb1[col];
            #pragma unroll
            for (int v = 0; v < 8; ++v)
                tb[(rb + v) * 72 + col] = (_Float16)sspf(acc[nt][v] + bb);
        }
        asm volatile("" ::: "memory");

        // ---- GEMM2: [16 x 64] x [64 x 64] ----
        v8f acc2[4] = {};
        #pragma unroll
        for (int ks = 0; ks < 2; ++ks) {
            const _Float16* rp = tb + n_lo * 72 + ks * 32;
            v8h lo = *(const v8h*)(rp + kbA);
            v8h hx = *(const v8h*)(rp + 16 + kbA);
            v16h a = make_a(lo, hx);
            #pragma unroll
            for (int nt = 0; nt < 4; ++nt) {
                v16h b = *(const v16h*)(fw2p + (size_t)((ks * 4 + nt) * 32 + lane) * 16);
                acc2[nt] = wmma16(a, b, acc2[nt]);
            }
        }
        asm volatile("" ::: "memory");

        // ---- ssp, gather h[src], multiply, atomic scatter to out[dst] ----
        if (e0 + 16 <= E && (E & 3) == 0) {      // full tile: vector index loads
            int sa[8], da[8];
            *(int4*)&sa[0] = *(const int4*)(eidx + e0 + rb);
            *(int4*)&sa[4] = *(const int4*)(eidx + e0 + rb + 4);
            *(int4*)&da[0] = *(const int4*)(eidx + E + e0 + rb);
            *(int4*)&da[4] = *(const int4*)(eidx + E + e0 + rb + 4);
            #pragma unroll
            for (int v = 0; v < 8; ++v) {
                const float* hr = h + (size_t)sa[v] * 64;
                float* orow = out + (size_t)da[v] * 64;
                #pragma unroll
                for (int nt = 0; nt < 4; ++nt) {
                    int col = nt * 16 + n_lo;
                    float f = sspf(acc2[nt][v] + fb2[col]);
                    atomicAdd(orow + col, f * hr[col]);
                }
            }
        } else {
            #pragma unroll
            for (int v = 0; v < 8; ++v) {
                int e = e0 + rb + v;
                if (e < E) {
                    int s = eidx[e];
                    int d = eidx[E + e];
                    const float* hr = h + (size_t)s * 64;
                    float* orow = out + (size_t)d * 64;
                    #pragma unroll
                    for (int nt = 0; nt < 4; ++nt) {
                        int col = nt * 16 + n_lo;
                        float f = sspf(acc2[nt][v] + fb2[col]);
                        atomicAdd(orow + col, f * hr[col]);
                    }
                }
            }
        }
    }
}

// ---------------------------------------------------------------------------
// out = ssp(agg@w2+b2)@w3 + b3 + x, where agg lives in `out`.
// ---------------------------------------------------------------------------
__global__ void __launch_bounds__(128)
node_out(const float* __restrict__ x, const _Float16* __restrict__ w2p,
         const float* __restrict__ b2, const _Float16* __restrict__ w3p,
         const float* __restrict__ b3, float* __restrict__ out, int N) {
    __shared__ _Float16 sT[4][16 * 72];
    int lane = threadIdx.x & 31, wv = threadIdx.x >> 5;
    int n_lo = lane & 15, hi = lane >> 4;
    int kbA = hi * 8, rb = hi * 8;
    _Float16* tb = sT[wv];
    int tiles = (N + 15) >> 4;
    for (int t = blockIdx.x * 4 + wv; t < tiles; t += gridDim.x * 4) {
        int r0 = t << 4;
        int rowc = min(r0 + n_lo, N - 1);
        const float* ar = out + (size_t)rowc * 64;

        v8f acc[4] = {};
        #pragma unroll
        for (int ks = 0; ks < 2; ++ks) {
            int kb = ks * 32 + kbA;
            float4 c0 = *(const float4*)(ar + kb);
            float4 c1 = *(const float4*)(ar + kb + 4);
            float4 c2 = *(const float4*)(ar + kb + 16);
            float4 c3 = *(const float4*)(ar + kb + 20);
            v16h a;
            a[0]=(_Float16)c0.x;  a[1]=(_Float16)c0.y;  a[2]=(_Float16)c0.z;  a[3]=(_Float16)c0.w;
            a[4]=(_Float16)c1.x;  a[5]=(_Float16)c1.y;  a[6]=(_Float16)c1.z;  a[7]=(_Float16)c1.w;
            a[8]=(_Float16)c2.x;  a[9]=(_Float16)c2.y;  a[10]=(_Float16)c2.z; a[11]=(_Float16)c2.w;
            a[12]=(_Float16)c3.x; a[13]=(_Float16)c3.y; a[14]=(_Float16)c3.z; a[15]=(_Float16)c3.w;
            #pragma unroll
            for (int nt = 0; nt < 4; ++nt) {
                v16h b = *(const v16h*)(w2p + (size_t)((ks * 4 + nt) * 32 + lane) * 16);
                acc[nt] = wmma16(a, b, acc[nt]);
            }
        }
        #pragma unroll
        for (int nt = 0; nt < 4; ++nt) {
            int col = nt * 16 + n_lo;
            float bb = b2[col];
            #pragma unroll
            for (int v = 0; v < 8; ++v)
                tb[(rb + v) * 72 + col] = (_Float16)sspf(acc[nt][v] + bb);
        }
        asm volatile("" ::: "memory");

        v8f acc2[4] = {};
        #pragma unroll
        for (int ks = 0; ks < 2; ++ks) {
            const _Float16* rp = tb + n_lo * 72 + ks * 32;
            v8h lo = *(const v8h*)(rp + kbA);
            v8h hx = *(const v8h*)(rp + 16 + kbA);
            v16h a = make_a(lo, hx);
            #pragma unroll
            for (int nt = 0; nt < 4; ++nt) {
                v16h b = *(const v16h*)(w3p + (size_t)((ks * 4 + nt) * 32 + lane) * 16);
                acc2[nt] = wmma16(a, b, acc2[nt]);
            }
        }
        asm volatile("" ::: "memory");

        if (r0 + 16 <= N) {                 // full tile: branch-free epilogue
            #pragma unroll
            for (int nt = 0; nt < 4; ++nt) {
                int col = nt * 16 + n_lo;
                float bb = b3[col];
                #pragma unroll
                for (int v = 0; v < 8; ++v) {
                    size_t rr = (size_t)(r0 + rb + v) * 64 + col;
                    out[rr] = acc2[nt][v] + bb + x[rr];
                }
            }
        } else {
            #pragma unroll
            for (int nt = 0; nt < 4; ++nt) {
                int col = nt * 16 + n_lo;
                float bb = b3[col];
                #pragma unroll
                for (int v = 0; v < 8; ++v) {
                    int rr = r0 + rb + v;
                    if (rr < N)
                        out[(size_t)rr * 64 + col] =
                            acc2[nt][v] + bb + x[(size_t)rr * 64 + col];
                }
            }
        }
        asm volatile("" ::: "memory");
    }
}

extern "C" void kernel_launch(void* const* d_in, const int* in_sizes, int n_in,
                              void* d_out, int out_size, void* d_ws, size_t ws_size,
                              hipStream_t stream) {
    const float* x    = (const float*)d_in[0];
    const int*   eidx = (const int*)  d_in[1];
    const float* rbf  = (const float*)d_in[2];
    const float* w1   = (const float*)d_in[3];
    const float* b1   = (const float*)d_in[4];
    const float* fw1  = (const float*)d_in[5];
    const float* fb1  = (const float*)d_in[6];
    const float* fw2  = (const float*)d_in[7];
    const float* fb2  = (const float*)d_in[8];
    const float* w2   = (const float*)d_in[9];
    const float* b2   = (const float*)d_in[10];
    const float* w3   = (const float*)d_in[11];
    const float* b3   = (const float*)d_in[12];
    const int N = in_sizes[0] / 64;
    const int E = in_sizes[1] / 2;
    float* out = (float*)d_out;

    // Workspace layout
    char* ws = (char*)d_ws;
    _Float16* fw1p = (_Float16*)ws; ws += 40960;   // 10 ksteps * 4 nt * 512 halves
    _Float16* fw2p = (_Float16*)ws; ws += 8192;
    _Float16* w1p  = (_Float16*)ws; ws += 8192;
    _Float16* w2p  = (_Float16*)ws; ws += 8192;
    _Float16* w3p  = (_Float16*)ws; ws += 8192;
    float* h = (float*)ws;                          // N*64 floats

    // 1) pack weights into f16 B-fragment layout
    pack_b<<<(40 * 512 + 255) / 256, 256, 0, stream>>>(fw1, fw1p, 300, 40);
    pack_b<<<(8 * 512 + 255) / 256, 256, 0, stream>>>(fw2, fw2p, 64, 8);
    pack_b<<<(8 * 512 + 255) / 256, 256, 0, stream>>>(w1,  w1p,  64, 8);
    pack_b<<<(8 * 512 + 255) / 256, 256, 0, stream>>>(w2,  w2p,  64, 8);
    pack_b<<<(8 * 512 + 255) / 256, 256, 0, stream>>>(w3,  w3p,  64, 8);

    // 2) zero the scatter accumulator (d_out doubles as agg buffer)
    int n4 = (N * 64) / 4;
    zero_f32<<<(n4 + 255) / 256, 256, 0, stream>>>(out, n4);

    // 3) h = x@w1 + b1
    int ntiles = (N + 15) / 16;
    node_linear<<<(ntiles + 3) / 4, 128, 0, stream>>>(x, w1p, b1, h, N);

    // 4) CFConv: filter net + gather*mul + atomic scatter
    int etiles = (E + 15) / 16;
    edge_cfconv<<<(etiles + 3) / 4, 128, 0, stream>>>(rbf, eidx, h, fw1p, fb1,
                                                      fw2p, fb2, out, E);

    // 5) out = ssp(agg@w2+b2)@w3 + b3 + x
    node_out<<<(ntiles + 3) / 4, 128, 0, stream>>>(x, w2p, b2, w3p, b3, out, N);
}
